// SparseCoupleDeConvTest_8942121910448
// MI455X (gfx1250) — matrix operand order repaired and checked by reading, not compiled, not run on hardware
//
#include <hip/hip_runtime.h>

typedef __attribute__((ext_vector_type(16))) _Float16 v16h;
typedef __attribute__((ext_vector_type(8)))  _Float16 h8;
typedef __attribute__((ext_vector_type(8)))  float    v8f;

#define HH 383
#define WW 383
#define CINC 64
#define COUTC 128
#define MD 191          // mid spatial size: (383-3)/2+1
#define NB 4
#define K1 576          // 9*64
#define K2 1152         // 9*128
#define NPIX (HH*WW)    // 146689
#define TILES_X 12      // ceil(191/16)
#define W1_LDS_HALVES (144*512)   // 18 k-chunks * 8 n-tiles, 512 halves/frag = 147456 B

static __device__ __forceinline__ v16h pack16(h8 lo, h8 hi) {
  return __builtin_shufflevector(lo, hi, 0,1,2,3,4,5,6,7,8,9,10,11,12,13,14,15);
}

// ---- transpose W2 to f16 K-major rows for WMMA B fragments ----
// w2t[cin][k], k = (ky*3+kx)*128 + cout (K2=1152)
__global__ void prep_weights(const float* __restrict__ W2, _Float16* __restrict__ w2t) {
  int t = blockIdx.x * blockDim.x + threadIdx.x;
  if (t < CINC * K2) {
    int c = t / K2, k2 = t % K2;
    w2t[(size_t)c * K2 + k2] = (_Float16)W2[(size_t)k2 * CINC + c];
  }
}

// ---- scatter sparse features into dense NHWC f16 grid ----
__global__ void scatter_feats(const float* __restrict__ feats, const int* __restrict__ coors,
                              _Float16* __restrict__ dense, int n) {
  int g = blockIdx.x * blockDim.x + threadIdx.x;
  int i = g >> 6, ch = g & 63;
  if (i >= n) return;
  int b = coors[3*i], y = coors[3*i+1], x = coors[3*i+2];
  dense[(((size_t)b*HH + y)*WW + x)*CINC + ch] = (_Float16)feats[(size_t)i*CINC + ch];
}

// ---- bin points by (y%2, x%2) parity class so deconv waves only run valid taps ----
__global__ void classify(const int* __restrict__ coors, int* __restrict__ cnt,
                         int* __restrict__ perm, int n) {
  int i = blockIdx.x * blockDim.x + threadIdx.x;
  if (i >= n) return;
  int y = coors[3*i+1], x = coors[3*i+2];
  int c = ((y & 1) << 1) | (x & 1);
  int r = atomicAdd(&cnt[c], 1);
  perm[(size_t)c * n + r] = i;
}

// ---- conv1: one wave = 16 mid positions x 128 couts, K = 9 taps * 64ch ----
// W1 staged once per block into LDS in fragment order:
//   frag f = chunk*8 + nt  (chunk = tap*2+cc, 18 chunks; nt = cout tile, 8)
//   halves [f*512 + lane*8 + e]       = lo half (frag VGPRs 0..3) for lane
//   halves [f*512 + 256 + lane*8 + e] = hi half (frag VGPRs 4..7) for lane
// so each ds_load_b128 is a contiguous, conflict-free 512B sweep across lanes.
__global__ __launch_bounds__(256) void conv1(const _Float16* __restrict__ dense,
                                             const float* __restrict__ W1,
                                             _Float16* __restrict__ mid) {
  extern __shared__ _Float16 wlds[];
  int tid = threadIdx.x;
  for (int L = tid; L < W1_LDS_HALVES; L += 256) {
    int f    = L >> 9;          // fragment id
    int r    = L & 511;
    int part = r >> 8;          // 0 = lo(v0..3), 1 = hi(v4..7)
    int r2   = r & 255;
    int lane = r2 >> 3;
    int e    = r2 & 7;
    int chunk = f >> 3;         // tap*2 + cc
    int nt    = f & 7;
    int row = nt * 16 + (lane & 15);                       // cout
    int k   = chunk * 32 + part * 16 + (lane >> 4) * 8 + e; // K index
    // W1 layout (ky,kx,cin,cout): flat = k*COUTC + row  (k = tap*64+cin)
    wlds[L] = (_Float16)W1[(size_t)k * COUTC + row];
  }
  __syncthreads();

  int wid = blockIdx.x * 8 + (tid >> 5);
  int ln  = tid & 31;
  if (wid >= NB * MD * TILES_X) return;
  int b   = wid / (MD * TILES_X);
  int r   = wid % (MD * TILES_X);
  int my  = r / TILES_X;
  int mx0 = (r % TILES_X) * 16;
  int m   = ln & 15;
  int hi  = ln >> 4;
  int mx  = mx0 + m; if (mx > MD-1) mx = MD-1;   // clamp: keep EXEC all-1s

  v8f acc[8] = {};
  const _Float16* lbase = wlds + ln * 8;
  for (int tap = 0; tap < 9; ++tap) {
    int ky = tap / 3, kx = tap % 3;
    const _Float16* apix =
        dense + (((size_t)b*HH + (2*my + ky))*WW + (2*mx + kx))*CINC + hi*8;
#pragma unroll
    for (int cc = 0; cc < 2; ++cc) {
      int c0 = cc * 32;
      v16h a = pack16(*(const h8*)(apix + c0), *(const h8*)(apix + c0 + 16));
      int fbase = (tap * 2 + cc) * 8;
#pragma unroll
      for (int nt = 0; nt < 8; ++nt) {
        const _Float16* fp = lbase + (size_t)(fbase + nt) * 512;
        v16h bf = pack16(*(const h8*)fp, *(const h8*)(fp + 256));
        acc[nt] = __builtin_amdgcn_wmma_f32_16x16x32_f16(
            false, a, false, bf, (short)0, acc[nt], false, false);
      }
    }
  }
  size_t mrow = ((size_t)b * MD + my) * MD;
#pragma unroll
  for (int v = 0; v < 8; ++v) {
    int mxs = mx0 + hi * 8 + v;
    if (mxs <= MD - 1) {
      _Float16* dst = mid + (mrow + mxs) * COUTC + m;
#pragma unroll
      for (int nt = 0; nt < 8; ++nt) dst[nt * 16] = (_Float16)acc[nt][v];
    }
  }
}

// ---- deconv: one wave = 16 sparse points (one parity class) x 64 channels ----
__global__ __launch_bounds__(256) void deconv(const _Float16* __restrict__ mid,
                                              const _Float16* __restrict__ w2t,
                                              const int* __restrict__ coors,
                                              const int* __restrict__ cnt,
                                              const int* __restrict__ perm,
                                              float* __restrict__ out, int n) {
  int wid = blockIdx.x * 8 + (threadIdx.x >> 5);
  int ln  = threadIdx.x & 31;
  int m   = ln & 15, hi = ln >> 4;
  int c0n = cnt[0], c1n = cnt[1], c2n = cnt[2], c3n = cnt[3];
  int t0 = (c0n + 15) >> 4, t1 = (c1n + 15) >> 4, t2 = (c2n + 15) >> 4, t3 = (c3n + 15) >> 4;
  int cls, tl, nc;
  if      (wid < t0)            { cls = 0; tl = wid;            nc = c0n; }
  else if (wid < t0+t1)         { cls = 1; tl = wid-t0;         nc = c1n; }
  else if (wid < t0+t1+t2)      { cls = 2; tl = wid-t0-t1;      nc = c2n; }
  else if (wid < t0+t1+t2+t3)   { cls = 3; tl = wid-t0-t1-t2;   nc = c3n; }
  else return;

  const int* permc = perm + (size_t)cls * n;
  int slotA = tl * 16 + m; if (slotA > nc - 1) slotA = nc - 1;   // duplicate tail point
  int ia = permc[slotA];
  int pb = coors[3*ia], py = coors[3*ia+1], px = coors[3*ia+2];
  int oy = cls >> 1, ox = cls & 1;
  int nky = oy ? 1 : 2, nkx = ox ? 1 : 2;

  v8f acc[4] = {};
  const v16h zero = {};
  const _Float16* w2row = w2t + (size_t)m * K2 + hi * 8;
  for (int ai = 0; ai < nky; ++ai) {
    int ky = oy ? 1 : ai * 2;
    int myv = (py - ky) >> 1;
    bool vy = (py >= ky) && (myv <= MD - 1);
    int myc = myv < 0 ? 0 : (myv > MD - 1 ? MD - 1 : myv);
    for (int bi = 0; bi < nkx; ++bi) {
      int kx = ox ? 1 : bi * 2;
      int mxv = (px - kx) >> 1;
      bool valid = vy && (px >= kx) && (mxv <= MD - 1);
      int mxc = mxv < 0 ? 0 : (mxv > MD - 1 ? MD - 1 : mxv);
      const _Float16* apix = mid + (((size_t)pb*MD + myc)*MD + mxc)*COUTC + hi*8;
      int tapk = (ky * 3 + kx) * COUTC;
#pragma unroll
      for (int cc = 0; cc < 4; ++cc) {
        int c0 = cc * 32;
        int k0 = tapk + c0;
        v16h a = pack16(*(const h8*)(apix + c0), *(const h8*)(apix + c0 + 16));
        a = valid ? a : zero;                    // per-lane zero, EXEC stays all-1s
#pragma unroll
        for (int nt = 0; nt < 4; ++nt) {
          const _Float16* bp = w2row + (size_t)nt * 16 * K2 + k0;
          v16h bf = pack16(*(const h8*)bp, *(const h8*)(bp + 16));
          acc[nt] = __builtin_amdgcn_wmma_f32_16x16x32_f16(
              false, a, false, bf, (short)0, acc[nt], false, false);
        }
      }
    }
  }
#pragma unroll
  for (int v = 0; v < 8; ++v) {
    int slot = tl * 16 + hi * 8 + v;
    if (slot < nc) {
      int idx = permc[slot];
      int sb = coors[3*idx], sy = coors[3*idx+1], sx = coors[3*idx+2];
      size_t ob = ((size_t)sb * CINC) * NPIX + (size_t)sy * WW + sx;   // NCHW
#pragma unroll
      for (int nt = 0; nt < 4; ++nt)
        out[ob + (size_t)(nt * 16 + m) * NPIX] = acc[nt][v];
    }
  }
}

extern "C" void kernel_launch(void* const* d_in, const int* in_sizes, int n_in,
                              void* d_out, int out_size, void* d_ws, size_t ws_size,
                              hipStream_t stream) {
  (void)n_in; (void)ws_size;
  const float* feats = (const float*)d_in[0];
  const int*   coors = (const int*)d_in[1];
  const float* W1    = (const float*)d_in[3];
  const float* W2    = (const float*)d_in[4];
  float* out = (float*)d_out;
  int n = in_sizes[0] / CINC;                 // 100000 points

  char* ws = (char*)d_ws;
  size_t o = 0;
  int* cnt  = (int*)(ws + o);      o += 256;
  int* perm = (int*)(ws + o);      o += (((size_t)4 * n * sizeof(int)) + 255) & ~(size_t)255;
  _Float16* w2t = (_Float16*)(ws + o); o += (size_t)CINC * K2 * 2;
  _Float16* dense = (_Float16*)(ws + o);
  size_t denseBytes = (size_t)NB * NPIX * CINC * 2;   o += denseBytes;
  _Float16* mid = (_Float16*)(ws + o);

  hipMemsetAsync(cnt, 0, 16, stream);
  hipMemsetAsync(dense, 0, denseBytes, stream);
  hipMemsetAsync(d_out, 0, (size_t)out_size * sizeof(float), stream);

  prep_weights<<<(CINC*K2 + 255) / 256, 256, 0, stream>>>(W2, w2t);
  scatter_feats<<<(n * 64 + 255) / 256, 256, 0, stream>>>(feats, coors, dense, n);
  classify<<<(n + 255) / 256, 256, 0, stream>>>(coors, cnt, perm, n);

  conv1<<<(NB * MD * TILES_X + 7) / 8, 256, W1_LDS_HALVES * 2, stream>>>(dense, W1, mid);

  int maxTiles = (n + 15) / 16 + 4;           // >= sum of per-class ceil tiles
  deconv<<<(maxTiles + 7) / 8, 256, 0, stream>>>(mid, w2t, coors, cnt, perm, out, n);
}
